// Transformer_72825465471425
// MI455X (gfx1250) — matrix operand order repaired and checked by reading, not compiled, not run on hardware
//
#include <hip/hip_runtime.h>
#include <hip/hip_bf16.h>

// ---------------------------------------------------------------------------
// GPT forward for gfx1250 (MI455X): all matmuls via v_wmma_f32_16x16x32_f16.
// B=4 T=1024 E=1024 H=16 HD=64 L=8 V=32000, BT=4096.
// ---------------------------------------------------------------------------

typedef float    v4f  __attribute__((ext_vector_type(4)));
typedef float    v8f  __attribute__((ext_vector_type(8)));
typedef _Float16 v4h  __attribute__((ext_vector_type(4)));
typedef _Float16 v8h  __attribute__((ext_vector_type(8)));
typedef _Float16 v16h __attribute__((ext_vector_type(16)));

static __device__ __forceinline__ v8h cvt8(v4f a, v4f b) {
  v4h ha = __builtin_convertvector(a, v4h);
  v4h hb = __builtin_convertvector(b, v4h);
  return __builtin_shufflevector(ha, hb, 0,1,2,3,4,5,6,7);
}

static __device__ __forceinline__ v16h pack16(v4f a, v4f b, v4f c, v4f d) {
  v8h lo = cvt8(a, b);
  v8h hi = cvt8(c, d);
  return __builtin_shufflevector(lo, hi, 0,1,2,3,4,5,6,7,8,9,10,11,12,13,14,15);
}

static __device__ __forceinline__ v16h cat16(v8h a, v8h b) {
  return __builtin_shufflevector(a, b, 0,1,2,3,4,5,6,7,8,9,10,11,12,13,14,15);
}

static __device__ __forceinline__ v8f wmma32(v16h a, v16h b, v8f c) {
  // (neg_a, A, neg_b, B, c_mod, C, reuse_a, reuse_b)
  return __builtin_amdgcn_wmma_f32_16x16x32_f16(false, a, false, b, (short)0, c,
                                                false, false);
}

// ---------------------------------------------------------------------------
// Embedding: x[bt,e] = tok_emb[inputs[bt],e] + pos_emb[t,e]   (v4f vectorized)
// ---------------------------------------------------------------------------
__global__ __launch_bounds__(256) void embed_kernel(const int* __restrict__ idx,
                                                    const float* __restrict__ tok,
                                                    const float* __restrict__ pos,
                                                    float* __restrict__ X) {
  const size_t i4 = ((size_t)blockIdx.x * 256 + threadIdx.x) * 4;  // over BT*E
  const int    e  = (int)(i4 & 1023);
  const size_t bt = i4 >> 10;
  const int    t  = (int)(bt & 1023);
  const v4f a = *(const v4f*)&tok[(size_t)idx[bt] * 1024 + e];
  const v4f b = *(const v4f*)&pos[(size_t)t * 1024 + e];
  *(v4f*)&X[i4] = a + b;
}

// ---------------------------------------------------------------------------
// LayerNorm over E=1024, one 256-thread block per row (v4f per thread).
// ---------------------------------------------------------------------------
__global__ __launch_bounds__(256) void ln_kernel(const float* __restrict__ X,
                                                 const float* __restrict__ g,
                                                 const float* __restrict__ b,
                                                 float* __restrict__ Y) {
  __shared__ float sm[256];
  const int    tid = threadIdx.x;
  const size_t row = blockIdx.x;
  const int    c0  = tid * 4;
  const v4f xv = *(const v4f*)&X[row * 1024 + c0];
  float s = xv.x + xv.y + xv.z + xv.w;
  sm[tid] = s; __syncthreads();
  for (int off = 128; off > 0; off >>= 1) {
    if (tid < off) sm[tid] += sm[tid + off];
    __syncthreads();
  }
  const float mean = sm[0] * (1.f / 1024.f);
  __syncthreads();
  v4f dv = xv - mean;
  float s2 = dv.x*dv.x + dv.y*dv.y + dv.z*dv.z + dv.w*dv.w;
  sm[tid] = s2; __syncthreads();
  for (int off = 128; off > 0; off >>= 1) {
    if (tid < off) sm[tid] += sm[tid + off];
    __syncthreads();
  }
  const float rstd = rsqrtf(sm[0] * (1.f / 1024.f) + 1e-5f);
  const v4f gv = *(const v4f*)&g[c0];
  const v4f bv = *(const v4f*)&b[c0];
  *(v4f*)&Y[row * 1024 + c0] = dv * rstd * gv + bv;
}

// ---------------------------------------------------------------------------
// Templated WMMA GEMM: Out[4096,N] = epilogue(A[4096,K] * B[K,N]).
//   BMODE 0: B row-major [K,N].  BMODE 1: B per-head [H][K][64], n=(h<<6)|d.
//   OMODE 0: Out row-major.      OMODE 1: scatter to [B,H,T,64] (QKV).
//   epilogue order: +bias -> relu -> +resid.  NT: nontemporal output stores.
// Block: 256 thr = 8 waves (2x4), tile 64(M) x 128(N), K-step 32,
// double-buffered LDS (one barrier per K-step), pipelined global loads.
// Compile-time N/K -> immediate-offset global loads, no 64-bit muls.
// ---------------------------------------------------------------------------
template <int N, int K, int BMODE, int OMODE, int RELU, int NT>
__global__ __launch_bounds__(256) void gemm_t(
    const float* __restrict__ A, const float* __restrict__ Bm,
    const float* __restrict__ bias, const float* __restrict__ resid,
    float* __restrict__ Out) {
  __shared__ __attribute__((aligned(16))) _Float16 As[2][64][40];
  __shared__ __attribute__((aligned(16))) _Float16 Bs[2][128][40];  // [n][k]

  const int tid  = threadIdx.x;
  const int lane = tid & 31;
  const int wave = tid >> 5;
  const int hi   = lane >> 4;
  const int lm   = lane & 15;
  const int wm   = (wave >> 2) * 32;
  const int wn   = (wave & 3) * 32;
  const int bm   = blockIdx.x * 64;
  const int bn   = blockIdx.y * 128;

  // fixed staging slice per thread
  const int ar = tid >> 2;                 // A row 0..63
  const int ac = (tid & 3) * 8;            // A col {0,8,16,24}
  const int bnn0 = tid & 127;              // B col
  const int bk0a = (tid >> 7) * 8;         // {0,8}
  const int bk0b = 16 + (tid >> 7) * 8;    // {16,24}

  // precomputed base pointers (all further offsets are compile-time)
  const float* aPtr = A + (size_t)(bm + ar) * K + ac;
  const float* bPtr;
  if (BMODE == 0) {
    bPtr = Bm + (size_t)bn + bnn0;                       // + (kk+k)*N
  } else {
    const int nn = bn + bnn0;
    bPtr = Bm + (size_t)(nn >> 6) * K * 64 + (nn & 63);  // + (kk+k)*64
  }

  v8f acc[2][2] = {};
  v4f a0, a1;
  float bv[2][8];

  auto load_tile = [&](int kk) {
    a0 = *(const v4f*)(aPtr + kk);
    a1 = *(const v4f*)(aPtr + kk + 4);
#pragma unroll
    for (int ch = 0; ch < 2; ++ch) {
      const int k0 = ch ? bk0b : bk0a;
#pragma unroll
      for (int k2 = 0; k2 < 8; ++k2) {
        const size_t stride = (BMODE == 0) ? (size_t)N : (size_t)64;
        bv[ch][k2] = bPtr[(size_t)(kk + k0 + k2) * stride];
      }
    }
  };

  auto commit = [&](int p) {  // staged regs -> LDS buffer p (fp32 -> f16)
    *(v8h*)&As[p][ar][ac] = cvt8(a0, a1);
#pragma unroll
    for (int ch = 0; ch < 2; ++ch) {
      const int k0 = ch ? bk0b : bk0a;
      v8h hv;
#pragma unroll
      for (int k2 = 0; k2 < 8; ++k2) hv[k2] = (_Float16)bv[ch][k2];
      *(v8h*)&Bs[p][bnn0][k0] = hv;
    }
  };

  auto compute = [&](int p) {
    // A elem e -> k = (e>>3)*16 + hi*8 + (e&7), row = lm   (contiguous 8s)
    // B elem e -> k = (e>>3)*16 + hi*8 + (e&7), col = lm    (same order as A)
    v16h a[2], b[2];
#pragma unroll
    for (int im = 0; im < 2; ++im) {
      const _Float16* rowp = &As[p][wm + im * 16 + lm][0];
      a[im] = cat16(*(const v8h*)(rowp + hi * 8),
                    *(const v8h*)(rowp + 16 + hi * 8));
    }
#pragma unroll
    for (int in = 0; in < 2; ++in) {
      const _Float16* colp = &Bs[p][wn + in * 16 + lm][0];
      b[in] = cat16(*(const v8h*)(colp + hi * 8),
                    *(const v8h*)(colp + 16 + hi * 8));
    }
#pragma unroll
    for (int im = 0; im < 2; ++im)
#pragma unroll
      for (int in = 0; in < 2; ++in)
        acc[im][in] = wmma32(a[im], b[in], acc[im][in]);
  };

  // prologue: tile 0 into buffer 0
  load_tile(0);
  commit(0);
  __syncthreads();

  // two K-steps per iteration; one barrier per K-step (double buffer)
  for (int kk = 0; kk < K; kk += 64) {
    load_tile(kk + 32);        // always valid: K % 64 == 0
    compute(0);
    commit(1);
    __syncthreads();
    if (kk + 64 < K) load_tile(kk + 64);
    compute(1);
    if (kk + 64 < K) {
      commit(0);
      __syncthreads();
    }
  }

  // epilogue; C/D layout: row = r + 8*hi, col = lm within each 16x16 tile
#pragma unroll
  for (int im = 0; im < 2; ++im)
#pragma unroll
    for (int in = 0; in < 2; ++in)
#pragma unroll
      for (int r = 0; r < 8; ++r) {
        const int row = bm + wm + im * 16 + r + hi * 8;
        const int col = bn + wn + in * 16 + lm;
        float v = acc[im][in][r];
        if (bias)   v += bias[col];
        if (RELU)   v = fmaxf(v, 0.f);
        if (resid)  v += resid[(size_t)row * N + col];
        if (OMODE == 0) {
          float* p = &Out[(size_t)row * N + col];
          if (NT) __builtin_nontemporal_store(v, p);
          else    *p = v;
        } else {  // scatter [bt, h*64+d] -> [B,H,T,64]
          const int bq = row >> 10, t = row & 1023;
          const int h = col >> 6, d = col & 63;
          Out[(((size_t)(bq * 16) + h) * 1024 + t) * 64 + d] = v;
        }
      }
}

// ---------------------------------------------------------------------------
// Fused causal attention (flash-style), reference quirk S = K * Q^T:
//   S[i,j] = K_i . Q_j (masked j<=i), A = softmax_j(S), O_i = sum_j A[i,j] V_j.
// One wave per (b*H+h, 16-row i-tile). O written as [B,T,H*64].
// With j0 stepping by 32 under the causal bound, max j touched is T-1,
// so no bounds guards are needed on Q/V loads.
// ---------------------------------------------------------------------------
__global__ __launch_bounds__(32) void attn_kernel(const float* __restrict__ Kg,
                                                  const float* __restrict__ Qg,
                                                  const float* __restrict__ Vg,
                                                  float* __restrict__ Og) {
  __shared__ __attribute__((aligned(16))) _Float16 P[16][40];
  const int lane = threadIdx.x;
  const int hi = lane >> 4, lm = lane & 15;
  const int bh = blockIdx.x;       // b*16 + h
  const int i0 = blockIdx.y * 16;  // i-tile origin
  const float* Kp = Kg + (size_t)bh * 1024 * 64;
  const float* Qp = Qg + (size_t)bh * 1024 * 64;
  const float* Vp = Vg + (size_t)bh * 1024 * 64;

  // A fragments of the 16 K-rows (fixed per block); contiguous 8-float chunks
  v16h aK[2];
#pragma unroll
  for (int kc = 0; kc < 2; ++kc) {
    const float* kp = &Kp[(size_t)(i0 + lm) * 64 + kc * 32];
    aK[kc] = pack16(*(const v4f*)(kp + hi * 8),
                    *(const v4f*)(kp + hi * 8 + 4),
                    *(const v4f*)(kp + 16 + hi * 8),
                    *(const v4f*)(kp + 16 + hi * 8 + 4));
  }

  v8f oacc[4] = {};
  float mrun[8], lsum[8];
#pragma unroll
  for (int r = 0; r < 8; ++r) { mrun[r] = -1e30f; lsum[r] = 0.f; }

  for (int j0 = 0; j0 <= i0 + 15; j0 += 32) {
    // S tiles for the two 16-wide j subtiles; Q fragment matches aK (8-d chunks)
    v8f s[2];
#pragma unroll
    for (int js = 0; js < 2; ++js) {
      v8f sacc = {};
      const int j = j0 + js * 16 + lm;
#pragma unroll
      for (int kc = 0; kc < 2; ++kc) {
        const float* qp = &Qp[(size_t)j * 64 + kc * 32 + hi * 8];
        const v16h bq = pack16(*(const v4f*)(qp + 0), *(const v4f*)(qp + 4),
                               *(const v4f*)(qp + 16), *(const v4f*)(qp + 20));
        sacc = wmma32(aK[kc], bq, sacc);
      }
      s[js] = sacc;
    }
    // scale + causal mask (C layout: row = r + 8*hi, col = lm)
#pragma unroll
    for (int js = 0; js < 2; ++js)
#pragma unroll
      for (int r = 0; r < 8; ++r) {
        const int irow = i0 + r + hi * 8;
        const int j = j0 + js * 16 + lm;
        const float v = s[js][r] * 0.125f;  // HD^-0.5
        s[js][r] = (j <= irow) ? v : -1e30f;
      }
    // online softmax; rows live in 16-lane halves -> butterfly over 4 bits
    float fr[8];
#pragma unroll
    for (int r = 0; r < 8; ++r) {
      float rmax = fmaxf(s[0][r], s[1][r]);
      rmax = fmaxf(rmax, __shfl_xor(rmax, 1, 32));
      rmax = fmaxf(rmax, __shfl_xor(rmax, 2, 32));
      rmax = fmaxf(rmax, __shfl_xor(rmax, 4, 32));
      rmax = fmaxf(rmax, __shfl_xor(rmax, 8, 32));
      const float mnew = fmaxf(mrun[r], rmax);
      const float f = __expf(mrun[r] - mnew);
      mrun[r] = mnew;
      const float p0 = __expf(s[0][r] - mnew);
      const float p1 = __expf(s[1][r] - mnew);
      float rs = p0 + p1;
      rs += __shfl_xor(rs, 1, 32);
      rs += __shfl_xor(rs, 2, 32);
      rs += __shfl_xor(rs, 4, 32);
      rs += __shfl_xor(rs, 8, 32);
      lsum[r] = lsum[r] * f + rs;
      s[0][r] = p0; s[1][r] = p1;
      fr[r] = f;
    }
#pragma unroll
    for (int dt = 0; dt < 4; ++dt)
#pragma unroll
      for (int r = 0; r < 8; ++r) oacc[dt][r] *= fr[r];

    // restage P (16x32) through LDS into A-fragment layout
    __syncthreads();
#pragma unroll
    for (int js = 0; js < 2; ++js)
#pragma unroll
      for (int r = 0; r < 8; ++r)
        P[r + hi * 8][js * 16 + lm] = (_Float16)s[js][r];
    __syncthreads();
    const v16h aP = cat16(*(const v8h*)&P[lm][hi * 16],
                          *(const v8h*)&P[lm][hi * 16 + 8]);

    // O += P @ V over d in 4 tiles of 16 (V loads coalesced across lanes)
#pragma unroll
    for (int dt = 0; dt < 4; ++dt) {
      v16h bvv;
#pragma unroll
      for (int e = 0; e < 16; ++e) {
        const int j = j0 + (hi << 4) + e;
        bvv[e] = (_Float16)Vp[(size_t)j * 64 + dt * 16 + lm];
      }
      oacc[dt] = wmma32(aP, bvv, oacc[dt]);
    }
  }

  // write O as [B, T, H*64]
  const int b = bh >> 4, h = bh & 15;
#pragma unroll
  for (int dt = 0; dt < 4; ++dt)
#pragma unroll
    for (int r = 0; r < 8; ++r) {
      const int irow = i0 + r + hi * 8;
      const int d = dt * 16 + lm;
      Og[((size_t)(b * 1024) + irow) * 1024 + h * 64 + d] = oacc[dt][r] / lsum[r];
    }
}

// ---------------------------------------------------------------------------
// Host orchestration
// ---------------------------------------------------------------------------
extern "C" void kernel_launch(void* const* d_in, const int* in_sizes, int n_in,
                              void* d_out, int out_size, void* d_ws, size_t ws_size,
                              hipStream_t stream) {
  (void)in_sizes; (void)n_in; (void)out_size;
  constexpr int Bb = 4, T = 1024, E = 1024, H = 16, L = 8, V = 32000;
  constexpr int BT = Bb * T;                 // 4096
  constexpr size_t S = (size_t)BT * E;       // 4,194,304 floats

  const int*   inputs  = (const int*)  d_in[0];
  const float* tok_emb = (const float*)d_in[1];
  const float* pos_emb = (const float*)d_in[2];
  const float* wk      = (const float*)d_in[3];
  const float* wq      = (const float*)d_in[4];
  const float* wv      = (const float*)d_in[5];
  const float* wo      = (const float*)d_in[6];
  const float* bo      = (const float*)d_in[7];
  const float* ln1_g   = (const float*)d_in[8];
  const float* ln1_b   = (const float*)d_in[9];
  const float* ln2_g   = (const float*)d_in[10];
  const float* ln2_b   = (const float*)d_in[11];
  const float* w1      = (const float*)d_in[12];
  const float* b1      = (const float*)d_in[13];
  const float* w2      = (const float*)d_in[14];
  const float* b2      = (const float*)d_in[15];
  const float* lnf_g   = (const float*)d_in[16];
  const float* lnf_b   = (const float*)d_in[17];
  const float* head_w  = (const float*)d_in[18];
  const float* head_b  = (const float*)d_in[19];
  float* out = (float*)d_out;

  if (ws_size < 11 * S * sizeof(float)) return;  // need ~176 MB scratch
  float* ws  = (float*)d_ws;
  float* xb  = ws + 0 * S;  // activations [BT,E]
  float* z1b = ws + 1 * S;  // LN1 output  [BT,E]
  float* qb  = ws + 2 * S;  // [B,H,T,64]
  float* kb  = ws + 3 * S;
  float* vb  = ws + 4 * S;
  float* ob  = ws + 5 * S;  // attention out [BT,E]
  float* z2b = ws + 6 * S;  // LN2 output  [BT,E]
  float* hb  = ws + 7 * S;  // MLP hidden  [BT,4E] (4*S)

  const dim3 blk256(256), blk32(32);
  const dim3 gEmb(BT * E / 1024);         // 4 floats per thread
  const dim3 gLN(BT);
  const dim3 gE(BT / 64, E / 128);        // N=1024
  const dim3 g4E(BT / 64, 4 * E / 128);   // N=4096
  const dim3 gV(BT / 64, V / 128);        // N=32000
  const dim3 gAttn(Bb * H, T / 16);

  embed_kernel<<<gEmb, blk256, 0, stream>>>(inputs, tok_emb, pos_emb, xb);

  for (int l = 0; l < L; ++l) {
    const float* wk_l = wk + (size_t)l * H * E * 64;
    const float* wq_l = wq + (size_t)l * H * E * 64;
    const float* wv_l = wv + (size_t)l * H * E * 64;
    const float* wo_l = wo + (size_t)l * E * E;
    const float* bo_l = bo + (size_t)l * E;
    const float* w1_l = w1 + (size_t)l * E * 4 * E;
    const float* b1_l = b1 + (size_t)l * 4 * E;
    const float* w2_l = w2 + (size_t)l * 4 * E * E;
    const float* b2_l = b2 + (size_t)l * E;

    // z1 = LN1(x)
    ln_kernel<<<gLN, blk256, 0, stream>>>(xb, ln1_g + (size_t)l * E,
                                          ln1_b + (size_t)l * E, z1b);
    // K,Q,V = z1 @ w{k,q,v}[h]  -> [B,H,T,64]
    gemm_t<1024, 1024, 1, 1, 0, 0><<<gE, blk256, 0, stream>>>(
        z1b, wk_l, nullptr, nullptr, kb);
    gemm_t<1024, 1024, 1, 1, 0, 0><<<gE, blk256, 0, stream>>>(
        z1b, wq_l, nullptr, nullptr, qb);
    gemm_t<1024, 1024, 1, 1, 0, 0><<<gE, blk256, 0, stream>>>(
        z1b, wv_l, nullptr, nullptr, vb);
    // causal attention (S = K Q^T), O -> [BT, E]
    attn_kernel<<<gAttn, blk32, 0, stream>>>(kb, qb, vb, ob);
    // x = z1 + O @ wo + bo   (residual from LN output, per reference)
    gemm_t<1024, 1024, 0, 0, 0, 0><<<gE, blk256, 0, stream>>>(
        ob, wo_l, bo_l, z1b, xb);
    // z2 = LN2(x)
    ln_kernel<<<gLN, blk256, 0, stream>>>(xb, ln2_g + (size_t)l * E,
                                          ln2_b + (size_t)l * E, z2b);
    // h = relu(z2 @ w1 + b1)
    gemm_t<4096, 1024, 0, 0, 1, 0><<<g4E, blk256, 0, stream>>>(
        z2b, w1_l, b1_l, nullptr, hb);
    // x = z2 + h @ w2 + b2
    gemm_t<1024, 4096, 0, 0, 0, 0><<<gE, blk256, 0, stream>>>(
        hb, w2_l, b2_l, z2b, xb);
  }

  // final LN + LM head (nontemporal logits stores: 524 MB, keep L2 clean)
  ln_kernel<<<gLN, blk256, 0, stream>>>(xb, lnf_g, lnf_b, z1b);
  gemm_t<32000, 1024, 0, 0, 0, 1><<<gV, blk256, 0, stream>>>(
      z1b, head_w, head_b, nullptr, out);
}
